// ResidualLSTMBlock_3401614098525
// MI455X (gfx1250) — matrix operand order repaired and checked by reading, not compiled, not run on hardware
//
#include <hip/hip_runtime.h>
#include <hip/hip_bf16.h>
#include <math.h>

// ---------------------------------------------------------------------------
// ResidualLSTMBlock for MI455X (gfx1250), wave32 + WMMA bf16 + async-to-LDS.
//
// Pipeline (all on `stream`, stream order gives dependencies):
//   0) convert weights fp32->bf16 (w_ih/w_hh as-is [N][K]; w1/w2 transposed to
//      n-major), bias = b_ih+b_hh, zero h ping-pong buffers
//   1) pre-LN(x) -> XN bf16                       (BW bound)
//   2) GX = XN @ w_ih^T + bias  (WMMA GEMM, out bf16, time-major [S][B][4H])
//   3) LSTM scan: persistent 16-workgroup cluster kernel, w_hh slice in LDS
//      (130 KB/WGP), c-state in WMMA accumulator VGPRs, cluster barrier/step
//   4) x2 = x + r ; HN = post-LN(x2) bf16
//   5) MLPH = gelu(HN @ w1 + b1)   (WMMA GEMM, bf16 out)
//   6) out  = x2 + MLPH @ w2 + b2  (WMMA GEMM, fp32 out)
// ---------------------------------------------------------------------------

typedef __bf16 bf16;
typedef __attribute__((ext_vector_type(16))) __bf16 v16bf;
typedef __attribute__((ext_vector_type(8)))  float  v8f;

#define WMMA_BF16(acc, a, b)                                                   \
  (acc) = __builtin_amdgcn_wmma_f32_16x16x32_bf16(false, (a), false, (b),      \
                                                  (short)0, (acc), false, false)

union Frag { uint4 q[2]; v16bf v; };

__device__ __forceinline__ v8f vzero8() {
  v8f z;
  for (int i = 0; i < 8; ++i) z[i] = 0.f;
  return z;
}

// Low 32 bits of a generic pointer into LDS == LDS byte address (aperture
// layout, ISA 10.2: addr[63:32] = shared aperture, addr[31:0] = LDS offset).
__device__ __forceinline__ unsigned lds_addr32(const void* p) {
  return (unsigned)(size_t)p;
}

// Async DMA: 16 bytes per lane, global -> LDS, tracked by ASYNCcnt (ISA 08).
__device__ __forceinline__ void async_copy_b128(unsigned lds, const void* gsrc) {
  asm volatile("global_load_async_to_lds_b128 %0, %1, off"
               :: "v"(lds), "v"(gsrc) : "memory");
}
__device__ __forceinline__ void wait_async0() {
  asm volatile("s_wait_asynccnt 0x0" ::: "memory");
}

__device__ __forceinline__ void cluster_barrier() {
#if __has_builtin(__builtin_amdgcn_s_cluster_barrier)
  __builtin_amdgcn_s_cluster_barrier();
#else
  asm volatile("s_barrier_signal -3\n\ts_barrier_wait -3" ::: "memory");
#endif
}

// A fragment (16x32 bf16, contiguous-K order used by this file): lane l holds row m0+(l&15);
// halves 0..15 = K kbase..kbase+15,
// kbase = 16*(l>=16).  `A` pre-offset to (m0, kc); lda in elements.
__device__ __forceinline__ v16bf load_a_frag(const bf16* A, int lda, int lane) {
  const bf16* p = A + (lane & 15) * lda + ((lane >> 4) << 4);
  Frag f;
  f.q[0] = *(const uint4*)(p);
  f.q[1] = *(const uint4*)(p + 8);
  return f.v;
}

// B fragment (32x16 bf16) from n-major weights Wt[N][K] (B[k][n]=Wt[n][k]):
// lane l holds column n0+(l&15); halves 0..15 = K kbase..kbase+15,
// kbase = 16*(l>=16).  `Wrow` pre-offset to Wt + n0*ldk + kc.
__device__ __forceinline__ v16bf load_b_frag(const bf16* Wrow, int ldk, int lane) {
  const bf16* p = Wrow + (lane & 15) * ldk + ((lane >> 4) << 4);
  Frag f;
  f.q[0] = *(const uint4*)(p);
  f.q[1] = *(const uint4*)(p + 8);
  return f.v;
}

// ---------------------------------------------------------------------------
// Prep kernels
// ---------------------------------------------------------------------------
__global__ void cvt_bf16_kernel(const float* __restrict__ s, bf16* __restrict__ d, int n) {
  for (int i = blockIdx.x * blockDim.x + threadIdx.x; i < n; i += gridDim.x * blockDim.x)
    d[i] = (bf16)s[i];
}

// s: [rows][cols] fp32 row-major; d: [cols][rows] bf16 (d[c*rows+r] = s[r*cols+c])
__global__ void transpose_bf16_kernel(const float* __restrict__ s, bf16* __restrict__ d,
                                      int rows, int cols) {
  int n = rows * cols;
  for (int i = blockIdx.x * blockDim.x + threadIdx.x; i < n; i += gridDim.x * blockDim.x) {
    int c = i / rows, r = i - c * rows;
    d[i] = (bf16)s[r * cols + c];
  }
}

__global__ void bias_sum_kernel(const float* __restrict__ a, const float* __restrict__ b,
                                float* __restrict__ d, int n) {
  for (int i = blockIdx.x * blockDim.x + threadIdx.x; i < n; i += gridDim.x * blockDim.x)
    d[i] = a[i] + b[i];
}

__global__ void zero_bf16_kernel(bf16* __restrict__ d, int n) {
  for (int i = blockIdx.x * blockDim.x + threadIdx.x; i < n; i += gridDim.x * blockDim.x)
    d[i] = (bf16)0.f;
}

// ---------------------------------------------------------------------------
// LayerNorm kernels: one wave32 per row of 512 floats (16 f32/lane)
// ---------------------------------------------------------------------------
__global__ __launch_bounds__(256) void pre_ln_kernel(
    const float* __restrict__ x, const float* __restrict__ g,
    const float* __restrict__ b, bf16* __restrict__ xn, int nrows) {
  int wave = threadIdx.x >> 5, lane = threadIdx.x & 31;
  int row = blockIdx.x * 8 + wave;
  if (row >= nrows) return;
  const float* xr = x + (size_t)row * 512;
  float4 v[4];
  float s = 0.f;
  for (int i = 0; i < 4; ++i) {
    v[i] = ((const float4*)xr)[lane + i * 32];
    s += v[i].x + v[i].y + v[i].z + v[i].w;
  }
  for (int off = 16; off > 0; off >>= 1) s += __shfl_xor(s, off, 32);
  float mu = s * (1.f / 512.f);
  float vs = 0.f;
  for (int i = 0; i < 4; ++i) {
    float dx = v[i].x - mu, dy = v[i].y - mu, dz = v[i].z - mu, dw = v[i].w - mu;
    vs += dx * dx + dy * dy + dz * dz + dw * dw;
  }
  for (int off = 16; off > 0; off >>= 1) vs += __shfl_xor(vs, off, 32);
  float rstd = rsqrtf(vs * (1.f / 512.f) + 1e-5f);
  bf16* xo = xn + (size_t)row * 512;
  for (int i = 0; i < 4; ++i) {
    int j = (lane + i * 32) * 4;
    xo[j + 0] = (bf16)((v[i].x - mu) * rstd * g[j + 0] + b[j + 0]);
    xo[j + 1] = (bf16)((v[i].y - mu) * rstd * g[j + 1] + b[j + 1]);
    xo[j + 2] = (bf16)((v[i].z - mu) * rstd * g[j + 2] + b[j + 2]);
    xo[j + 3] = (bf16)((v[i].w - mu) * rstd * g[j + 3] + b[j + 3]);
  }
}

// x2 = x + r (fp32 out), hn = LN(x2)*g+b (bf16 out)
__global__ __launch_bounds__(256) void residual_post_ln_kernel(
    const float* __restrict__ x, const float* __restrict__ r,
    const float* __restrict__ g, const float* __restrict__ b,
    float* __restrict__ x2, bf16* __restrict__ hn, int nrows) {
  int wave = threadIdx.x >> 5, lane = threadIdx.x & 31;
  int row = blockIdx.x * 8 + wave;
  if (row >= nrows) return;
  const float4* xr = (const float4*)(x + (size_t)row * 512);
  const float4* rr = (const float4*)(r + (size_t)row * 512);
  float4* x2r = (float4*)(x2 + (size_t)row * 512);
  float4 v[4];
  float s = 0.f;
  for (int i = 0; i < 4; ++i) {
    float4 a = xr[lane + i * 32], c = rr[lane + i * 32];
    v[i] = make_float4(a.x + c.x, a.y + c.y, a.z + c.z, a.w + c.w);
    x2r[lane + i * 32] = v[i];
    s += v[i].x + v[i].y + v[i].z + v[i].w;
  }
  for (int off = 16; off > 0; off >>= 1) s += __shfl_xor(s, off, 32);
  float mu = s * (1.f / 512.f);
  float vs = 0.f;
  for (int i = 0; i < 4; ++i) {
    float dx = v[i].x - mu, dy = v[i].y - mu, dz = v[i].z - mu, dw = v[i].w - mu;
    vs += dx * dx + dy * dy + dz * dz + dw * dw;
  }
  for (int off = 16; off > 0; off >>= 1) vs += __shfl_xor(vs, off, 32);
  float rstd = rsqrtf(vs * (1.f / 512.f) + 1e-5f);
  bf16* ho = hn + (size_t)row * 512;
  for (int i = 0; i < 4; ++i) {
    int j = (lane + i * 32) * 4;
    ho[j + 0] = (bf16)((v[i].x - mu) * rstd * g[j + 0] + b[j + 0]);
    ho[j + 1] = (bf16)((v[i].y - mu) * rstd * g[j + 1] + b[j + 1]);
    ho[j + 2] = (bf16)((v[i].z - mu) * rstd * g[j + 2] + b[j + 2]);
    ho[j + 3] = (bf16)((v[i].w - mu) * rstd * g[j + 3] + b[j + 3]);
  }
}

// ---------------------------------------------------------------------------
// WMMA GEMM: C[M,N] = A[M,K] * Wt^T + bias, epilogue per MODE.
// Block = 256 threads (8 waves), tile 128x128, wave computes 64x32 (4x2 tiles).
// A tile DMA'd to LDS with global_load_async_to_lds_b128, double-buffered:
// slab k+1 streams in (ASYNCcnt) while WMMAs consume slab k; one barrier/slab.
// Pad stride 40 -> conflict-free ds_load_b128.
// MODE 0: out bf16, GX time-major [S][B][4H]   (m = b*1024 + t)
// MODE 1: out bf16 = gelu(acc + bias), [M,N]
// MODE 2: out fp32 = acc + bias + addsrc, [M,N]
// ---------------------------------------------------------------------------
template <int MODE>
__global__ __launch_bounds__(256) void wmma_gemm_kernel(
    const bf16* __restrict__ A, const bf16* __restrict__ Wt,
    const float* __restrict__ bias, void* __restrict__ out_p,
    const float* __restrict__ addsrc, int M, int N, int K) {
  __shared__ bf16 As[2][128 * 40];
  const int lane = threadIdx.x & 31;
  const int wave = threadIdx.x >> 5;
  const int wm = wave & 1;   // 0..1 -> M offset wm*64
  const int wn = wave >> 1;  // 0..3 -> N offset wn*32
  const int M0 = blockIdx.y * 128;
  const int N0 = blockIdx.x * 128;

  v8f acc[4][2];
  for (int i = 0; i < 4; ++i)
    for (int j = 0; j < 2; ++j) acc[i][j] = vzero8();

  const int row = threadIdx.x >> 1;   // 0..127
  const int half = threadIdx.x & 1;   // 32B chunk within the 64B row

  // issue async DMA of A[M0..M0+127][kc..kc+31] into buffer `buf`
  auto issue_slab = [&](int buf, int kc) {
    const bf16* gsrc = A + (size_t)(M0 + row) * K + kc + half * 16;
    unsigned lds = lds_addr32(&As[buf][row * 40 + half * 16]);
    async_copy_b128(lds, gsrc);
    async_copy_b128(lds + 16u, gsrc + 8);
  };

  const int nslab = K >> 5;
  issue_slab(0, 0);

  for (int s = 0; s < nslab; ++s) {
    const int kc = s << 5;
    const int ib = s & 1;
    wait_async0();       // my DMA writes into As[ib] have landed
    __syncthreads();     // everyone's writes landed; prev readers of As[ib^1] done
    if (s + 1 < nslab) issue_slab(ib ^ 1, kc + 32);

    v16bf bfrag[2];
    for (int j = 0; j < 2; ++j) {
      int n0 = N0 + wn * 32 + j * 16;
      bfrag[j] = load_b_frag(Wt + (size_t)n0 * K + kc, K, lane);
    }
    if (s + 1 < nslab)   // keep next weight slab warm (global_prefetch_b8)
      __builtin_prefetch(Wt + (size_t)(N0 + wn * 32 + lane) * K + kc + 32, 0, 0);

    v16bf afrag[4];
    for (int i = 0; i < 4; ++i)
      afrag[i] = load_a_frag(&As[ib][(wm * 64 + i * 16) * 40], 40, lane);

    for (int i = 0; i < 4; ++i)
      for (int j = 0; j < 2; ++j) WMMA_BF16(acc[i][j], afrag[i], bfrag[j]);
  }

  // Epilogue; C layout: elem k of v8f -> m = k + 8*(lane>=16), n = lane&15
  for (int i = 0; i < 4; ++i)
    for (int j = 0; j < 2; ++j) {
      int mbase = M0 + wm * 64 + i * 16 + ((lane >> 4) << 3);
      int n = N0 + wn * 32 + j * 16 + (lane & 15);
      float bv = bias[n];
      for (int k = 0; k < 8; ++k) {
        int m = mbase + k;
        float v = acc[i][j][k] + bv;
        if (MODE == 0) {
          int bidx = m >> 10, t = m & 1023;  // m = b*1024 + t
          ((bf16*)out_p)[((size_t)t * 64 + bidx) * 2048 + n] = (bf16)v;
        } else if (MODE == 1) {
          float gl = 0.5f * v * (1.0f + erff(v * 0.70710678118654752f));
          ((bf16*)out_p)[(size_t)m * N + n] = (bf16)gl;
        } else {
          ((float*)out_p)[(size_t)m * N + n] = v + addsrc[(size_t)m * N + n];
        }
      }
    }
}

// ---------------------------------------------------------------------------
// LSTM scan: 16 workgroups (one cluster; each on its own WGP), 256 thr each.
// Block bk owns hidden slice j0=bk*32. LDS holds w_hh rows {g*512+j0..+31}
// for all 4 gates: 4*32*520*2B = 130 KB (row stride 520 -> conflict-free),
// DMA'd in with async-to-LDS. Wave (mt=wave&3, nt=wave>>2) computes a
// 16(batch)x16(hidden) block: 4 gate accumulators + persistent c-state in
// VGPRs. Per step: 64 WMMAs/wave, gate nonlinearity, write h (bf16 ping-pong,
// L2-resident) + r (fp32), then device fence + cluster barrier.
// ---------------------------------------------------------------------------
__global__ __launch_bounds__(256) void lstm_seq_kernel(
    const bf16* __restrict__ gx,   // [S][B][2048] bf16, time-major
    const bf16* __restrict__ whh,  // [2048][512] bf16 (n-major)
    bf16* __restrict__ h0buf, bf16* __restrict__ h1buf,  // [64][512]
    float* __restrict__ r) {       // [B][S][512]
  extern __shared__ bf16 smem[];   // 4*32*520 bf16
  const int LDW = 520;
  const int lane = threadIdx.x & 31;
  const int wave = threadIdx.x >> 5;
  const int mt = wave & 3;   // batch tile: rows mt*16..mt*16+15
  const int nt = wave >> 2;  // hidden subtile: cols nt*16..nt*16+15 of 32
  const int j0 = blockIdx.x * 32;

  // DMA w_hh slice into LDS (32 x b128 per thread, ASYNCcnt <= 32 < 63)
  for (int idx = threadIdx.x; idx < 4 * 32 * 64; idx += 256) {
    int chunk = idx & 63;          // 64 chunks of 8 bf16 = 512
    int nl = (idx >> 6) & 31;
    int gate = idx >> 11;
    const bf16* src = whh + (size_t)(gate * 512 + j0 + nl) * 512 + chunk * 8;
    async_copy_b128(lds_addr32(smem + (gate * 32 + nl) * LDW + chunk * 8), src);
  }
  wait_async0();
  __syncthreads();

  v8f cst = vzero8();  // cell state, register-resident for all 1024 steps
  const bf16* hprev = h0buf;
  bf16* hnext = h1buf;

  const int mbase = mt * 16 + ((lane >> 4) << 3);
  const int n = j0 + nt * 16 + (lane & 15);  // global hidden index

  for (int t = 0; t < 1024; ++t) {
    v8f acc[4];
    for (int g = 0; g < 4; ++g) acc[g] = vzero8();

    for (int kc = 0; kc < 512; kc += 32) {
      v16bf af = load_a_frag(hprev + (size_t)(mt * 16) * 512 + kc, 512, lane);
      for (int g = 0; g < 4; ++g) {
        v16bf bfrag = load_b_frag(smem + (g * 32 + nt * 16) * LDW + kc, LDW, lane);
        WMMA_BF16(acc[g], af, bfrag);
      }
    }

    const bf16* gxt = gx + (size_t)t * 64 * 2048;
    for (int k = 0; k < 8; ++k) {
      int bidx = mbase + k;  // batch index
      float gi = acc[0][k] + (float)gxt[(size_t)bidx * 2048 + 0 + n];
      float gf = acc[1][k] + (float)gxt[(size_t)bidx * 2048 + 512 + n];
      float gg = acc[2][k] + (float)gxt[(size_t)bidx * 2048 + 1024 + n];
      float go = acc[3][k] + (float)gxt[(size_t)bidx * 2048 + 1536 + n];
      float i_ = 1.f / (1.f + __expf(-gi));
      float f_ = 1.f / (1.f + __expf(-gf));
      float o_ = 1.f / (1.f + __expf(-go));
      float cn = f_ * cst[k] + i_ * tanhf(gg);
      cst[k] = cn;
      float h = o_ * tanhf(cn);
      hnext[(size_t)bidx * 512 + n] = (bf16)h;
      r[((size_t)bidx * 1024 + t) * 512 + n] = h;
    }

    __threadfence();     // make h visible at device scope
    __syncthreads();     // all waves in WG arrived
    cluster_barrier();   // sync the 16-WG cluster per step
    const bf16* tmp = hprev;
    hprev = hnext;
    hnext = (bf16*)tmp;
  }
}

// ---------------------------------------------------------------------------
// Host launcher
// ---------------------------------------------------------------------------
extern "C" void kernel_launch(void* const* d_in, const int* in_sizes, int n_in,
                              void* d_out, int out_size, void* d_ws, size_t ws_size,
                              hipStream_t stream) {
  (void)in_sizes; (void)n_in; (void)out_size; (void)ws_size;
  const int B = 64, S = 1024, H = 512;
  const int M = B * S;           // 65536
  const int FH = 4 * H;          // 2048

  const float* x      = (const float*)d_in[0];
  const float* pre_g  = (const float*)d_in[1];
  const float* pre_b  = (const float*)d_in[2];
  const float* w_ih   = (const float*)d_in[3];
  const float* w_hh   = (const float*)d_in[4];
  const float* b_ih   = (const float*)d_in[5];
  const float* b_hh   = (const float*)d_in[6];
  const float* post_g = (const float*)d_in[7];
  const float* post_b = (const float*)d_in[8];
  const float* w1     = (const float*)d_in[9];
  const float* b1     = (const float*)d_in[10];
  const float* w2     = (const float*)d_in[11];
  const float* b2     = (const float*)d_in[12];
  float* out = (float*)d_out;

  // workspace carve-out
  size_t off = 0;
  auto alloc = [&](size_t bytes) -> void* {
    void* p = (char*)d_ws + off;
    off += (bytes + 255) & ~(size_t)255;
    return p;
  };
  bf16*  XN   = (bf16*)alloc((size_t)M * H * 2);          // pre-LN(x) bf16
  bf16*  GX   = (bf16*)alloc((size_t)S * B * FH * 2);     // gates, time-major
  bf16*  WIH  = (bf16*)alloc((size_t)FH * H * 2);
  bf16*  WHH  = (bf16*)alloc((size_t)FH * H * 2);
  bf16*  W1T  = (bf16*)alloc((size_t)(2 * H) * H * 2);    // [2H][H] n-major
  bf16*  W2T  = (bf16*)alloc((size_t)H * (2 * H) * 2);    // [H][2H] n-major
  float* BIAS = (float*)alloc((size_t)FH * 4);
  bf16*  H0   = (bf16*)alloc((size_t)B * H * 2);
  bf16*  H1   = (bf16*)alloc((size_t)B * H * 2);
  float* R    = (float*)alloc((size_t)M * H * 4);         // lstm output
  float* X2   = (float*)alloc((size_t)M * H * 4);         // x + r
  bf16*  HN   = (bf16*)alloc((size_t)M * H * 2);          // post-LN bf16
  bf16*  MLPH = (bf16*)alloc((size_t)M * (2 * H) * 2);    // gelu(h@w1+b1)

  // 0) weight prep
  cvt_bf16_kernel<<<4096, 256, 0, stream>>>(w_ih, WIH, FH * H);
  cvt_bf16_kernel<<<4096, 256, 0, stream>>>(w_hh, WHH, FH * H);
  transpose_bf16_kernel<<<2048, 256, 0, stream>>>(w1, W1T, H, 2 * H);      // d[n*H+k]=w1[k][n]
  transpose_bf16_kernel<<<2048, 256, 0, stream>>>(w2, W2T, 2 * H, H);      // d[n*2H+k]=w2[k][n]
  bias_sum_kernel<<<8, 256, 0, stream>>>(b_ih, b_hh, BIAS, FH);
  zero_bf16_kernel<<<128, 256, 0, stream>>>(H0, B * H);
  zero_bf16_kernel<<<128, 256, 0, stream>>>(H1, B * H);

  // 1) pre-LN
  pre_ln_kernel<<<M / 8, 256, 0, stream>>>(x, pre_g, pre_b, XN, M);

  // 2) GX = XN @ w_ih^T + (b_ih + b_hh)   [M x 2048], K = 512
  wmma_gemm_kernel<0><<<dim3(FH / 128, M / 128), 256, 0, stream>>>(
      XN, WIH, BIAS, (void*)GX, nullptr, M, FH, H);

  // 3) recurrent scan: 16 WGs = one cluster, 130 KB dynamic LDS each
  lstm_seq_kernel<<<16, 256, 4 * 32 * 520 * sizeof(bf16), stream>>>(
      GX, WHH, H0, H1, R);

  // 4) x2 = x + r ; HN = post-LN(x2)
  residual_post_ln_kernel<<<M / 8, 256, 0, stream>>>(x, R, post_g, post_b, X2, HN, M);

  // 5) MLPH = gelu(HN @ w1 + b1)   [M x 1024], K = 512
  wmma_gemm_kernel<1><<<dim3((2 * H) / 128, M / 128), 256, 0, stream>>>(
      HN, W1T, b1, (void*)MLPH, nullptr, M, 2 * H, H);

  // 6) out = x2 + MLPH @ w2 + b2   [M x 512], K = 1024
  wmma_gemm_kernel<2><<<dim3(H / 128, M / 128), 256, 0, stream>>>(
      MLPH, W2T, b2, (void*)out, X2, M, H, 2 * H);
}
